// Model_58402965291291
// MI455X (gfx1250) — matrix operand order repaired and checked by reading, not compiled, not run on hardware
//
#include <hip/hip_runtime.h>

typedef __attribute__((ext_vector_type(16))) _Float16 v16h;
typedef __attribute__((ext_vector_type(8)))  float    v8f;

#define Dn    64
#define Kn    128
#define Wn    20
#define Hn    128
#define ROWS  16          // batch rows per block
#define F4    256         // 4*D
#define F4P   260         // padded row stride (bank-conflict free A-frag loads)

__device__ __forceinline__ float wred_sum(float v) {
#pragma unroll
    for (int m = 16; m; m >>= 1) v += __shfl_xor(v, m, 32);
    return v;
}

__global__ __launch_bounds__(256) void tgn_fused_kernel(
    const float* __restrict__ adj_time,
    const float* __restrict__ gc,
    const float* __restrict__ cur_time,
    const float* __restrict__ neigh_mask,
    const float* __restrict__ hist_feat,
    const float* __restrict__ hist_time,
    const float* __restrict__ t2v_w,
    const float* __restrict__ t2v_b,
    const float* __restrict__ node_w,
    const float* __restrict__ node_b,
    const float* __restrict__ att_w,
    const float* __restrict__ att_b,
    const float* __restrict__ weight,
    const int*   __restrict__ targets,
    const int*   __restrict__ neigh_idx,
    float* __restrict__ out)
{
    __shared__ float sT2w[Dn], sT2b[Dn], sNw[Dn], sNb[Dn], sAw[2 * Dn];
    __shared__ float sDtR[Kn], sGR[Kn], sMR[Kn];   // per-row neighbor gathers
    __shared__ float sNgh[Kn][Dn];                 // normalized neighbor encodings (32 KB)
    __shared__ float sScore[Kn];
    __shared__ float sPacc[4][Dn];
    __shared__ float sF4[ROWS][F4P];               // [hist_agg | tgt_feat | neigh_agg]
    __shared__ float sTgtAtt[ROWS];

    const int tid   = threadIdx.x;
    const int wave  = tid >> 5;
    const int lane  = tid & 31;
    const int bbase = blockIdx.x * ROWS;
    const float t    = cur_time[0];
    const float attb = att_b[0];

    // ---- Phase 0: stage parameter vectors in LDS ----
    if (tid < Dn) {
        sT2w[tid] = t2v_w[tid];  sT2b[tid] = t2v_b[tid];
        sNw[tid]  = node_w[tid]; sNb[tid]  = node_b[tid];
    }
    if (tid < 2 * Dn) sAw[tid] = att_w[tid];
    __syncthreads();

    // ---- Phase 1: target encodings (wave w -> rows 2w, 2w+1) ----
#pragma unroll
    for (int rr = 0; rr < 2; ++rr) {
        int   r   = wave * 2 + rr;
        int   idx = targets[bbase + r];
        float dt  = fabsf(t - adj_time[idx]);
        float g   = gc[idx];
        int d0 = lane, d1 = lane + 32;
        float p0  = sT2w[d0] * dt + sT2b[d0];
        float p1  = sT2w[d1] * dt + sT2b[d1];
        float tv0 = (d0 == 0) ? p0 : __cosf(p0);
        float tv1 = __cosf(p1);
        float f0  = fmaxf(tv0 + g * sNw[d0] + sNb[d0], 0.f);
        float f1  = fmaxf(tv1 + g * sNw[d1] + sNb[d1], 0.f);
        float ss  = wred_sum(f0 * f0 + f1 * f1);
        float rn  = 1.f / fmaxf(sqrtf(ss), 1e-12f);
        f0 *= rn;  f1 *= rn;
        sF4[r][128 + d0] = f0;
        sF4[r][128 + d1] = f1;
        float ad = wred_sum(f0 * sAw[d0] + f1 * sAw[d1]);
        if (lane == 0) sTgtAtt[r] = ad;
    }

    // ---- Phase 2: history aggregation (streams hist_feat once, coalesced) ----
#pragma unroll
    for (int rr = 0; rr < 2; ++rr) {
        int r = wave * 2 + rr;
        const float* hf = hist_feat + (size_t)(bbase + r) * Wn * (2 * Dn);
        const float* ht = hist_time + (size_t)(bbase + r) * Wn;
        float a0 = 0.f, a1 = 0.f, a2 = 0.f, a3 = 0.f;
        for (int w = 0; w < Wn; ++w) {
            float hw = 1.f / (2.f * (1.f + (t - ht[w])));
            const float* row = hf + w * (2 * Dn);
            a0 += hw * row[lane];
            a1 += hw * row[lane + 32];
            a2 += hw * row[lane + 64];
            a3 += hw * row[lane + 96];
        }
        sF4[r][lane]      = a0;
        sF4[r][lane + 32] = a1;
        sF4[r][lane + 64] = a2;
        sF4[r][lane + 96] = a3;
    }
    __syncthreads();

    // ---- Phase 3: per-row neighbor encode + attention + aggregate ----
    for (int r = 0; r < ROWS; ++r) {
        // gather this row's 128 neighbors
        if (tid < Kn) {
            int idx   = neigh_idx[(size_t)(bbase + r) * Kn + tid];
            sDtR[tid] = t - adj_time[idx];
            sGR[tid]  = gc[idx];
            sMR[tid]  = neigh_mask[(size_t)(bbase + r) * Kn + tid];
        }
        __syncthreads();

        const float tgtA = sTgtAtt[r];
#pragma unroll 1
        for (int i = 0; i < 16; ++i) {
            int   k   = wave * 16 + i;
            float dtr = sDtR[k];
            float g   = sGR[k];
            float dt  = fabsf(dtr);
            int d0 = lane, d1 = lane + 32;
            float p0  = sT2w[d0] * dt + sT2b[d0];
            float p1  = sT2w[d1] * dt + sT2b[d1];
            float tv0 = (d0 == 0) ? p0 : __cosf(p0);
            float tv1 = __cosf(p1);
            float f0  = fmaxf(tv0 + g * sNw[d0] + sNb[d0], 0.f);
            float f1  = fmaxf(tv1 + g * sNw[d1] + sNb[d1], 0.f);
            float ss  = wred_sum(f0 * f0 + f1 * f1);
            float rn  = 1.f / fmaxf(sqrtf(ss), 1e-12f);
            f0 *= rn;  f1 *= rn;
            sNgh[k][d0] = f0;
            sNgh[k][d1] = f1;
            float ad = wred_sum(f0 * sAw[Dn + d0] + f1 * sAw[Dn + d1]);
            if (lane == 0) {
                float ts = 1.f / (2.f * __logf(2.718281828459045f + dtr));
                float x  = ts + tgtA + ad + attb;
                float sc = (x > 0.f) ? x : 0.01f * x;
                sScore[k] = sc * sMR[k];
            }
        }
        __syncthreads();

        // neigh_agg: 256 threads, d = tid&63, k-chunk = tid>>6
        {
            int d = tid & 63, kc = tid >> 6;
            float acc = 0.f;
#pragma unroll
            for (int kk = 0; kk < 32; ++kk) {
                int k = kc * 32 + kk;
                acc += sScore[k] * sNgh[k][d];
            }
            sPacc[kc][d] = acc;
        }
        __syncthreads();
        if (tid < 64)
            sF4[r][192 + tid] = sPacc[0][tid] + sPacc[1][tid] + sPacc[2][tid] + sPacc[3][tid];
        __syncthreads();
    }

    // ---- Phase 4: out = relu(feat4 @ W^T) via v_wmma_f32_16x16x32_f16 ----
    // wave w owns output h-tile [16w, 16w+16); K = 256 in 8 chunks of 32.
    {
        const int hbase   = wave * 16;
        const int halfSel = lane >> 4;   // 0: lanes 0-15, 1: lanes 16-31
        const int l16     = lane & 15;
        v8f c = {};
#pragma unroll
        for (int kb = 0; kb < 8; ++kb) {
            int kbase = kb * 32;
            v16h a, b;
            // A frag: M=l16; elems 0-15 = K(kbase + 16*halfSel + i)
            {
                const float* a0 = &sF4[l16][kbase + (halfSel ? 16 : 0)];
                const float* a1 = &sF4[l16][kbase + (halfSel ? 24 : 8)];
#pragma unroll
                for (int i = 0; i < 8; ++i) a[i]     = (_Float16)a0[i];
#pragma unroll
                for (int i = 0; i < 8; ++i) a[8 + i] = (_Float16)a1[i];
            }
            // B frag: N=l16 (=> h = hbase+l16); elems 0-15 = K(kbase + 16*halfSel + i); B[k][n] = W[h][k]
            {
                const float* br = weight + (size_t)(hbase + l16) * F4 + kbase + (halfSel ? 16 : 0);
#pragma unroll
                for (int i = 0; i < 16; ++i) b[i] = (_Float16)br[i];
            }
            c = __builtin_amdgcn_wmma_f32_16x16x32_f16(false, a, false, b, (short)0, c, false, false);
        }
        // C/D layout: VGPR i -> M = i + 8*halfSel, N = l16
#pragma unroll
        for (int i = 0; i < 8; ++i) {
            int m = i + 8 * halfSel;
            out[(size_t)(bbase + m) * Hn + (hbase + l16)] = fmaxf(c[i], 0.f);
        }
    }
}

extern "C" void kernel_launch(void* const* d_in, const int* in_sizes, int n_in,
                              void* d_out, int out_size, void* d_ws, size_t ws_size,
                              hipStream_t stream) {
    (void)in_sizes; (void)n_in; (void)d_ws; (void)ws_size; (void)out_size;
    const float* adj_time   = (const float*)d_in[0];
    const float* gc         = (const float*)d_in[1];
    const float* cur_time   = (const float*)d_in[2];
    const float* neigh_mask = (const float*)d_in[3];
    const float* hist_feat  = (const float*)d_in[4];
    const float* hist_time  = (const float*)d_in[5];
    const float* t2v_w      = (const float*)d_in[6];
    const float* t2v_b      = (const float*)d_in[7];
    const float* node_w     = (const float*)d_in[8];
    const float* node_b     = (const float*)d_in[9];
    const float* att_w      = (const float*)d_in[10];
    const float* att_b      = (const float*)d_in[11];
    const float* weight     = (const float*)d_in[12];
    const int*   targets    = (const int*)d_in[13];
    const int*   neigh_idx  = (const int*)d_in[14];
    float* out = (float*)d_out;

    const int B = 4096;
    dim3 grid(B / ROWS), block(256);
    tgn_fused_kernel<<<grid, block, 0, stream>>>(
        adj_time, gc, cur_time, neigh_mask, hist_feat, hist_time,
        t2v_w, t2v_b, node_w, node_b, att_w, att_b, weight,
        targets, neigh_idx, out);
}
